// ChineseEcommerceMoE_88424786690746
// MI455X (gfx1250) — matrix operand order repaired and checked by reading, not compiled, not run on hardware
//
#include <hip/hip_runtime.h>
#include <hip/hip_bf16.h>

// ---------------- model constants ----------------
constexpr int V_  = 32000;
constexpr int D_  = 768;
constexpr int L_  = 4;
constexpr int H_  = 12;
constexpr int HD_ = 64;
constexpr int FF_ = 2048;
constexpr int E_  = 8;
constexpr int B_  = 2;
constexpr int S_  = 512;
constexpr int T_  = B_ * S_;          // 1024 tokens
constexpr float EPS_   = 1e-6f;
constexpr float SCALE_ = 0.125f;      // 64^-0.5

// ---------------- WMMA types ----------------
typedef __attribute__((ext_vector_type(16))) __bf16 v16bf;
typedef __attribute__((ext_vector_type(8)))  float  v8f;

union BF16x16 {
    v16bf v;
    unsigned int u[8];
    uint4 q[2];
};

// truncating fp32->bf16 (top 16 bits); compiler can use ds_store_b16_d16_hi
__device__ __forceinline__ unsigned short f2bf_trunc(float f) {
    return (unsigned short)(__float_as_uint(f) >> 16);
}
// pack two fp32 -> bf16x2 dword with a single v_perm_b32
__device__ __forceinline__ unsigned int pack_bf16x2(float lo, float hi) {
    return __builtin_amdgcn_perm(__float_as_uint(hi), __float_as_uint(lo), 0x07060302u);
}

#define LDSK 72   // 64 + 8 pad; 144 B row stride (16B aligned)

// stage 16 contiguous fp32 -> 16 bf16 in LDS (two uint4 stores)
__device__ __forceinline__ void stage_row16(const float* __restrict__ gsrc,
                                            unsigned short* __restrict__ ldst) {
    const float4* s4 = (const float4*)gsrc;
    #pragma unroll
    for (int j = 0; j < 2; ++j) {
        float4 p = s4[2 * j];
        float4 r = s4[2 * j + 1];
        uint4 w;
        w.x = pack_bf16x2(p.x, p.y);
        w.y = pack_bf16x2(p.z, p.w);
        w.z = pack_bf16x2(r.x, r.y);
        w.w = pack_bf16x2(r.z, r.w);
        *(uint4*)(ldst + j * 8) = w;
    }
}

#define WMMA_BF16(A, Bf, Cacc) \
    __builtin_amdgcn_wmma_f32_16x16x32_bf16(false, (A).v, false, (Bf).v, (short)0, (Cacc), false, false)

// load A fragment (16x32, ISA layout) from an LDS row of LDSK ushorts
__device__ __forceinline__ BF16x16 load_afrag(const unsigned short* row, int ks, int hi) {
    const uint4* r4 = (const uint4*)row;
    BF16x16 f;
    f.q[0] = r4[ks * 4 + hi];
    f.q[1] = r4[ks * 4 + 2 + hi];
    return f;
}
// load B fragment (32x16) from a transposed [n][k] LDS row
__device__ __forceinline__ BF16x16 load_bfrag(const unsigned short* row, int ks, int hi) {
    const uint4* r4 = (const uint4*)row;
    BF16x16 f;
    f.q[0] = r4[ks * 4 + hi];
    f.q[1] = r4[ks * 4 + 2 + hi];
    return f;
}

// ---------------- bf16 WMMA GEMM, software-pipelined ----------------
// C[M,N] (+)= A[M,K] * B[K,N]; REQUIRES M%128==0, N%64==0, K%64==0.
#define TILE_M 128
#define TILE_N 64
#define TILE_K 64

__global__ __launch_bounds__(256)
void gemm_bf16_wmma(const float* __restrict__ A, const float* __restrict__ B,
                    float* __restrict__ C,
                    int M, int N, int K,
                    int lda, int ldb, int ldc,
                    int accumulate)
{
    __shared__ __align__(16) unsigned short sA[TILE_M * LDSK]; // [m][k]
    __shared__ __align__(16) unsigned short sB[TILE_N * LDSK]; // [n][k] transposed

    const int tid  = threadIdx.x;
    const int lane = tid & 31;
    const int wave = tid >> 5;             // 0..7
    const int m0 = blockIdx.y * TILE_M;
    const int n0 = blockIdx.x * TILE_N;

    const int mt = wave;                   // 16-row strip per wave
    const int hi = lane >> 4;
    const int l15 = lane & 15;

    v8f acc0 = {}, acc1 = {}, acc2 = {}, acc3 = {};

    const int ra = tid >> 1;               // 0..127  A row
    const int ca = (tid & 1) * 32;         // A col base (32 floats)
    const int rb = tid >> 2;               // 0..63   B k-row
    const int cb = (tid & 3) * 16;         // B n-col base (16 floats)

    const float* agp_base = A + (long)(m0 + ra) * lda + ca;
    const float* bgp_base = B + (long)rb * ldb + n0 + cb;

    const unsigned short* sArow = sA + (mt * 16 + l15) * LDSK;
    unsigned short* sAst = sA + ra * LDSK + ca;

    // ---- software pipeline: registers hold the in-flight K-tile ----
    float4 ar[8];   // 32 floats of A
    float4 br[4];   // 16 floats of B
    {
        const float4* ag = (const float4*)agp_base;
        #pragma unroll
        for (int j = 0; j < 8; ++j) ar[j] = ag[j];
        const float4* bg = (const float4*)bgp_base;
        #pragma unroll
        for (int j = 0; j < 4; ++j) br[j] = bg[j];
    }

    for (int k0 = 0; k0 < K; k0 += TILE_K) {
        // ---- pack current regs -> LDS (v_perm + b128 / b16_hi stores) ----
        #pragma unroll
        for (int j = 0; j < 4; ++j) {
            float4 p = ar[2 * j];
            float4 r = ar[2 * j + 1];
            uint4 w;
            w.x = pack_bf16x2(p.x, p.y);
            w.y = pack_bf16x2(p.z, p.w);
            w.z = pack_bf16x2(r.x, r.y);
            w.w = pack_bf16x2(r.z, r.w);
            *(uint4*)(sAst + j * 8) = w;
        }
        #pragma unroll
        for (int j = 0; j < 4; ++j) {
            float4 p = br[j];
            const int nn = cb + 4 * j;
            sB[(nn + 0) * LDSK + rb] = f2bf_trunc(p.x);
            sB[(nn + 1) * LDSK + rb] = f2bf_trunc(p.y);
            sB[(nn + 2) * LDSK + rb] = f2bf_trunc(p.z);
            sB[(nn + 3) * LDSK + rb] = f2bf_trunc(p.w);
        }
        __syncthreads();

        // ---- issue next tile's global loads; latency hides behind WMMAs ----
        if (k0 + TILE_K < K) {
            const float4* ag = (const float4*)(agp_base + k0 + TILE_K);
            #pragma unroll
            for (int j = 0; j < 8; ++j) ar[j] = ag[j];
            const float4* bg = (const float4*)(bgp_base + (long)(k0 + TILE_K) * ldb);
            #pragma unroll
            for (int j = 0; j < 4; ++j) br[j] = bg[j];
        }

        // ---- 8 WMMAs from LDS ----
        #pragma unroll
        for (int ks = 0; ks < 2; ++ks) {
            BF16x16 af = load_afrag(sArow, ks, hi);
            BF16x16 b0 = load_bfrag(sB + (0 * 16 + l15) * LDSK, ks, hi);
            BF16x16 b1 = load_bfrag(sB + (1 * 16 + l15) * LDSK, ks, hi);
            BF16x16 b2 = load_bfrag(sB + (2 * 16 + l15) * LDSK, ks, hi);
            BF16x16 b3 = load_bfrag(sB + (3 * 16 + l15) * LDSK, ks, hi);
            acc0 = WMMA_BF16(af, b0, acc0);
            acc1 = WMMA_BF16(af, b1, acc1);
            acc2 = WMMA_BF16(af, b2, acc2);
            acc3 = WMMA_BF16(af, b3, acc3);
        }
        __syncthreads();
    }

    const int grow = m0 + mt * 16 + hi * 8;
    float* crow0 = C + (long)grow * ldc + n0 + l15;
    #pragma unroll
    for (int r = 0; r < 8; ++r) {
        float* cr = crow0 + (long)r * ldc;
        if (accumulate) {
            cr[0]  += acc0[r];
            cr[16] += acc1[r];
            cr[32] += acc2[r];
            cr[48] += acc3[r];
        } else {
            cr[0]  = acc0[r];
            cr[16] = acc1[r];
            cr[32] = acc2[r];
            cr[48] = acc3[r];
        }
    }
}

// ---------------- flash-style WMMA attention ----------------
// grid: (S/64, B*H); 256 threads. 64 queries per block; key blocks of 64.
#define AQ 64

__global__ __launch_bounds__(256)
void attn_wmma_kernel(const float* __restrict__ q, const float* __restrict__ k,
                      const float* __restrict__ v, float* __restrict__ o)
{
    const int q0 = blockIdx.x * AQ;
    const int h  = blockIdx.y % H_;
    const int b  = blockIdx.y / H_;

    __shared__ __align__(16) unsigned short sQ[AQ * LDSK];    // [q][d]
    __shared__ __align__(16) unsigned short sK[AQ * LDSK];    // [key][d] (B layout)
    __shared__ __align__(16) unsigned short sVt[HD_ * LDSK];  // [d][key] (B layout)
    __shared__ __align__(16) unsigned short sP[AQ * LDSK];    // [q][key] (A layout)
    __shared__ float sS[AQ * 68];                             // fp32 scores
    __shared__ float sM[AQ], sL[AQ], sC[AQ];

    const int tid  = threadIdx.x;
    const int lane = tid & 31;
    const int wave = tid >> 5;
    const int hi   = lane >> 4;
    const int l15  = lane & 15;
    const int mt   = wave & 3;             // q subtile 0..3
    const int nt0  = (wave >> 2) * 2;      // 0 or 2
    const int nt1  = nt0 + 1;
    const int srow = mt * 16 + hi * 8;

    const int sr   = tid >> 2;             // staging row 0..63
    const int scol = (tid & 3) * 16;       // staging col base

    const float* qbase = q + ((long)(b * S_) * D_) + h * HD_;
    const float* kbase = k + ((long)(b * S_) * D_) + h * HD_;
    const float* vbase = v + ((long)(b * S_) * D_) + h * HD_;

    // stage Q tile once
    stage_row16(qbase + (long)(q0 + sr) * D_ + scol, sQ + sr * LDSK + scol);
    if (tid < AQ) { sM[tid] = -1e30f; sL[tid] = 0.f; }

    v8f o0 = {}, o1 = {};

    for (int kb = 0; kb < S_ / AQ; ++kb) {
        const int kk0 = kb * AQ;
        // stage K block [key][d]
        stage_row16(kbase + (long)(kk0 + sr) * D_ + scol, sK + sr * LDSK + scol);
        // stage V block transposed [d][key]
        {
            const float4* s4 = (const float4*)(vbase + (long)(kk0 + sr) * D_ + scol);
            #pragma unroll
            for (int j = 0; j < 4; ++j) {
                float4 p = s4[j];
                const int dd = scol + 4 * j;
                sVt[(dd + 0) * LDSK + sr] = f2bf_trunc(p.x);
                sVt[(dd + 1) * LDSK + sr] = f2bf_trunc(p.y);
                sVt[(dd + 2) * LDSK + sr] = f2bf_trunc(p.z);
                sVt[(dd + 3) * LDSK + sr] = f2bf_trunc(p.w);
            }
        }
        __syncthreads();

        // S_blk = Q * K^T  (contraction over d = 64, two K=32 substeps)
        v8f s0 = {}, s1 = {};
        #pragma unroll
        for (int ks = 0; ks < 2; ++ks) {
            BF16x16 af = load_afrag(sQ + (mt * 16 + l15) * LDSK, ks, hi);
            BF16x16 b0 = load_bfrag(sK + (nt0 * 16 + l15) * LDSK, ks, hi);
            BF16x16 b1 = load_bfrag(sK + (nt1 * 16 + l15) * LDSK, ks, hi);
            s0 = WMMA_BF16(af, b0, s0);
            s1 = WMMA_BF16(af, b1, s1);
        }
        #pragma unroll
        for (int r = 0; r < 8; ++r) {
            sS[(srow + r) * 68 + nt0 * 16 + l15] = s0[r] * SCALE_;
            sS[(srow + r) * 68 + nt1 * 16 + l15] = s1[r] * SCALE_;
        }
        __syncthreads();

        // online softmax: 4 lanes per row, 16 cols each
        {
            const int row = tid >> 2;
            const int sub = tid & 3;
            float bm = -1e30f;
            #pragma unroll
            for (int i = 0; i < 16; ++i) bm = fmaxf(bm, sS[row * 68 + sub * 16 + i]);
            bm = fmaxf(bm, __shfl_xor(bm, 1));
            bm = fmaxf(bm, __shfl_xor(bm, 2));
            const float mold = sM[row];
            const float mnew = fmaxf(mold, bm);
            float psum = 0.f;
            #pragma unroll
            for (int i = 0; i < 16; ++i) {
                const int c = sub * 16 + i;
                const float p = __expf(sS[row * 68 + c] - mnew);
                sP[row * LDSK + c] = f2bf_trunc(p);
                psum += p;
            }
            psum += __shfl_xor(psum, 1);
            psum += __shfl_xor(psum, 2);
            if (sub == 0) {
                const float corr = __expf(mold - mnew);
                sM[row] = mnew;
                sL[row] = sL[row] * corr + psum;
                sC[row] = corr;
            }
        }
        __syncthreads();

        // O = O*corr + P * V   (contraction over 64 keys)
        #pragma unroll
        for (int r = 0; r < 8; ++r) {
            const float cc = sC[srow + r];
            o0[r] *= cc;
            o1[r] *= cc;
        }
        #pragma unroll
        for (int ks = 0; ks < 2; ++ks) {
            BF16x16 af = load_afrag(sP + (mt * 16 + l15) * LDSK, ks, hi);
            BF16x16 b0 = load_bfrag(sVt + (nt0 * 16 + l15) * LDSK, ks, hi);
            BF16x16 b1 = load_bfrag(sVt + (nt1 * 16 + l15) * LDSK, ks, hi);
            o0 = WMMA_BF16(af, b0, o0);
            o1 = WMMA_BF16(af, b1, o1);
        }
        __syncthreads();
    }

    // final normalize + store
    #pragma unroll
    for (int r = 0; r < 8; ++r) {
        const float inv = 1.0f / sL[srow + r];
        float* op = o + ((long)(b * S_ + q0 + srow + r) * D_) + h * HD_;
        op[nt0 * 16 + l15] = o0[r] * inv;
        op[nt1 * 16 + l15] = o1[r] * inv;
    }
}

// ---------------- embedding ----------------
__global__ __launch_bounds__(256)
void embed_kernel(const int* __restrict__ ids, const float* __restrict__ tok,
                  const float* __restrict__ pos, float* __restrict__ x)
{
    const int t = blockIdx.x;
    const int s = t % S_;
    const int id = ids[t];
    for (int i = threadIdx.x; i < D_; i += 256)
        x[(long)t * D_ + i] = tok[(long)id * D_ + i] + pos[(long)s * D_ + i];
}

// ---------------- RMSNorm ----------------
__global__ __launch_bounds__(256)
void rmsnorm_kernel(const float* __restrict__ x, const float* __restrict__ w,
                    float* __restrict__ y)
{
    const int t = blockIdx.x;
    const float* xr = x + (long)t * D_;
    __shared__ float red[256];
    float s = 0.f;
    for (int i = threadIdx.x; i < D_; i += 256) { float v = xr[i]; s += v * v; }
    red[threadIdx.x] = s;
    __syncthreads();
    for (int st = 128; st > 0; st >>= 1) {
        if (threadIdx.x < st) red[threadIdx.x] += red[threadIdx.x + st];
        __syncthreads();
    }
    const float inv = rsqrtf(red[0] / (float)D_ + EPS_);
    for (int i = threadIdx.x; i < D_; i += 256)
        y[(long)t * D_ + i] = w[i] * xr[i] * inv;
}

// ---------------- router logits: [T,E] = xn[T,D] @ rw[D,E] ----------------
__global__ __launch_bounds__(256)
void router_kernel(const float* __restrict__ xn, const float* __restrict__ rw,
                   float* __restrict__ logits)
{
    const int t = blockIdx.x;
    __shared__ float red[256 * E_];
    float acc[E_];
    #pragma unroll
    for (int e = 0; e < E_; ++e) acc[e] = 0.f;
    for (int i = threadIdx.x; i < D_; i += 256) {
        const float xv = xn[(long)t * D_ + i];
        #pragma unroll
        for (int e = 0; e < E_; ++e) acc[e] += xv * rw[(long)i * E_ + e];
    }
    #pragma unroll
    for (int e = 0; e < E_; ++e) red[threadIdx.x * E_ + e] = acc[e];
    __syncthreads();
    for (int st = 128; st > 0; st >>= 1) {
        if (threadIdx.x < st) {
            #pragma unroll
            for (int e = 0; e < E_; ++e)
                red[threadIdx.x * E_ + e] += red[(threadIdx.x + st) * E_ + e];
        }
        __syncthreads();
    }
    if (threadIdx.x < E_) logits[(long)t * E_ + threadIdx.x] = red[threadIdx.x];
}

// ---------------- router softmax + top-2 combine ----------------
__global__ __launch_bounds__(256)
void topk_kernel(const float* __restrict__ logits, float* __restrict__ combine)
{
    const int t = blockIdx.x * blockDim.x + threadIdx.x;
    if (t >= T_) return;
    float l[E_];
    float m = -1e30f;
    #pragma unroll
    for (int e = 0; e < E_; ++e) { l[e] = logits[t * E_ + e]; m = fmaxf(m, l[e]); }
    float s = 0.f;
    #pragma unroll
    for (int e = 0; e < E_; ++e) { l[e] = __expf(l[e] - m); s += l[e]; }
    #pragma unroll
    for (int e = 0; e < E_; ++e) l[e] /= s;
    int i0 = 0;
    #pragma unroll
    for (int e = 1; e < E_; ++e) if (l[e] > l[i0]) i0 = e;
    int i1 = (i0 == 0) ? 1 : 0;
    #pragma unroll
    for (int e = 0; e < E_; ++e) if (e != i0 && l[e] > l[i1]) i1 = e;
    const float tot = l[i0] + l[i1];
    #pragma unroll
    for (int e = 0; e < E_; ++e) combine[t * E_ + e] = 0.f;
    combine[t * E_ + i0] = l[i0] / tot;
    combine[t * E_ + i1] = l[i1] / tot;
}

// ---------------- silu(g) * u ----------------
__global__ __launch_bounds__(256)
void silu_mul_kernel(float* __restrict__ g, const float* __restrict__ u, int n)
{
    const int i = blockIdx.x * blockDim.x + threadIdx.x;
    if (i < n) {
        const float gv = g[i];
        const float sg = gv / (1.f + __expf(-gv));
        g[i] = sg * u[i];
    }
}

// ---------------- x += combine[:,e] * eo ----------------
__global__ __launch_bounds__(256)
void moe_acc_kernel(float* __restrict__ x, const float* __restrict__ eo,
                    const float* __restrict__ combine, int e)
{
    const int i = blockIdx.x * blockDim.x + threadIdx.x;
    if (i < T_ * D_) {
        const int t = i / D_;
        const float w = combine[t * E_ + e];
        x[i] += w * eo[i];
    }
}

// ---------------- host orchestration ----------------
static inline void launch_gemm(const float* A, const float* B, float* C,
                               int M, int N, int K, int accumulate,
                               hipStream_t stream)
{
    dim3 grid(N / TILE_N, M / TILE_M, 1);
    gemm_bf16_wmma<<<grid, 256, 0, stream>>>(A, B, C, M, N, K, K, N, N, accumulate);
}

extern "C" void kernel_launch(void* const* d_in, const int* in_sizes, int n_in,
                              void* d_out, int out_size, void* d_ws, size_t ws_size,
                              hipStream_t stream)
{
    (void)in_sizes; (void)n_in; (void)out_size; (void)ws_size;

    const int*   ids        = (const int*)  d_in[0];
    const float* embed_tok  = (const float*)d_in[1];
    const float* embed_pos  = (const float*)d_in[2];
    const float* wq         = (const float*)d_in[3];
    const float* wk         = (const float*)d_in[4];
    const float* wv         = (const float*)d_in[5];
    const float* wo         = (const float*)d_in[6];
    const float* norm1_w    = (const float*)d_in[7];
    const float* norm2_w    = (const float*)d_in[8];
    const float* router_w   = (const float*)d_in[9];
    const float* gate_w     = (const float*)d_in[10];
    const float* up_w       = (const float*)d_in[11];
    const float* down_w     = (const float*)d_in[12];
    const float* final_norm = (const float*)d_in[13];
    const float* lm_head    = (const float*)d_in[14];
    float*       out        = (float*)d_out;

    const long TD  = (long)T_ * D_;
    const long TFF = (long)T_ * FF_;
    const long TE  = (long)T_ * E_;

    float* ws      = (float*)d_ws;
    float* x       = ws;            // [T,D]
    float* xn      = x + TD;        // [T,D]
    float* q       = xn + TD;       // [T,D]
    float* k       = q + TD;        // [T,D]
    float* v       = k + TD;        // [T,D]
    float* attno   = v + TD;        // [T,D]
    float* eo      = attno + TD;    // [T,D]
    float* g       = eo + TD;       // [T,FF]
    float* u       = g + TFF;       // [T,FF]
    float* logits  = u + TFF;       // [T,E]
    float* combine = logits + TE;   // [T,E]

    embed_kernel<<<T_, 256, 0, stream>>>(ids, embed_tok, embed_pos, x);

    for (int l = 0; l < L_; ++l) {
        const float* wq_l = wq + (long)l * D_ * D_;
        const float* wk_l = wk + (long)l * D_ * D_;
        const float* wv_l = wv + (long)l * D_ * D_;
        const float* wo_l = wo + (long)l * D_ * D_;
        const float* rw_l = router_w + (long)l * D_ * E_;

        // ---- attention block ----
        rmsnorm_kernel<<<T_, 256, 0, stream>>>(x, norm1_w + (long)l * D_, xn);
        launch_gemm(xn, wq_l, q, T_, D_, D_, 0, stream);
        launch_gemm(xn, wk_l, k, T_, D_, D_, 0, stream);
        launch_gemm(xn, wv_l, v, T_, D_, D_, 0, stream);
        {
            dim3 agrid(S_ / AQ, B_ * H_, 1);
            attn_wmma_kernel<<<agrid, 256, 0, stream>>>(q, k, v, attno);
        }
        launch_gemm(attno, wo_l, x, T_, D_, D_, 1, stream);   // x += attno @ wo

        // ---- MoE block ----
        rmsnorm_kernel<<<T_, 256, 0, stream>>>(x, norm2_w + (long)l * D_, xn);
        router_kernel<<<T_, 256, 0, stream>>>(xn, rw_l, logits);
        topk_kernel<<<(T_ + 255) / 256, 256, 0, stream>>>(logits, combine);

        for (int e = 0; e < E_; ++e) {
            const float* gw_le = gate_w + ((long)l * E_ + e) * (long)D_ * FF_;
            const float* uw_le = up_w   + ((long)l * E_ + e) * (long)D_ * FF_;
            const float* dw_le = down_w + ((long)l * E_ + e) * (long)FF_ * D_;
            launch_gemm(xn, gw_le, g, T_, FF_, D_, 0, stream);
            launch_gemm(xn, uw_le, u, T_, FF_, D_, 0, stream);
            silu_mul_kernel<<<(int)((TFF + 255) / 256), 256, 0, stream>>>(g, u, (int)TFF);
            launch_gemm(g, dw_le, eo, T_, D_, FF_, 0, stream);
            moe_acc_kernel<<<(int)((TD + 255) / 256), 256, 0, stream>>>(x, eo, combine, e);
        }
    }

    // ---- final norm + lm head ----
    rmsnorm_kernel<<<T_, 256, 0, stream>>>(x, final_norm, xn);
    launch_gemm(xn, lm_head, out, T_, V_, D_, 0, stream);
}
